// GraphAttention_21397527068976
// MI455X (gfx1250) — matrix operand order repaired and checked by reading, not compiled, not run on hardware
//
#include <hip/hip_runtime.h>

typedef __attribute__((ext_vector_type(16))) _Float16 v16h;
typedef __attribute__((ext_vector_type(8)))  float    v8f;

namespace {
constexpr int   kB = 4;
constexpr int   kN = 4096;
constexpr int   kF = 128;
constexpr int   kU = 64;
constexpr float kNegBig = -9.0e15f;
constexpr float kBnEps  = 1.0e-3f;
}

// ---------------------------------------------------------------------------
// Kernel 1: Wh = h @ W, stored directly in WMMA B-operand order (f16):
//   whB[((b*(N/32) + jb)*4 + cg)*512 + lane*16 + k]
//   where for Wh row n = jb*32 + jrow, col u = cg*16 + c:
//     lane = (jrow >= 16 ? 16 : 0) + c,  k = jrow & 15
// (contiguous k order, same as kernel 2's A gather: lanes 0-15 hold K=0..15,
//  lanes 16-31 hold K=16..31, column = lane&15, half index = K - base).
// Also computes src = Wh@a1, dst = Wh@a2.
// ---------------------------------------------------------------------------
__global__ void __launch_bounds__(256)
gat_wh_src_dst_kernel(const float* __restrict__ h,
                      const float* __restrict__ W,
                      const float* __restrict__ a,
                      _Float16* __restrict__ whB,
                      float* __restrict__ srcOut,
                      float* __restrict__ dstOut)
{
  __shared__ float sW[kF * kU];       // 32 KB: full 128x64 W
  __shared__ float sRedS[64 * 4];
  __shared__ float sRedD[64 * 4];

  const int t = threadIdx.x;
  const size_t rowBase = (size_t)blockIdx.x * 64;

  for (int i = t; i < kF * kU; i += 256) sW[i] = W[i];
  __syncthreads();

  const int r  = t >> 2;            // local row 0..63
  const int cg = t & 3;             // U column group
  const int cb = cg * 16;           // col base

  float acc[16];
#pragma unroll
  for (int c = 0; c < 16; ++c) acc[c] = 0.f;

  const float4* h4 = (const float4*)(h + (rowBase + r) * kF);
#pragma unroll 4
  for (int k4 = 0; k4 < kF / 4; ++k4) {
    const float4 hv = h4[k4];
    const int k = k4 * 4;
#pragma unroll
    for (int c = 0; c < 16; ++c) {
      float v = acc[c];
      v = fmaf(hv.x, sW[(k + 0) * kU + cb + c], v);
      v = fmaf(hv.y, sW[(k + 1) * kU + cb + c], v);
      v = fmaf(hv.z, sW[(k + 2) * kU + cb + c], v);
      v = fmaf(hv.w, sW[(k + 3) * kU + cb + c], v);
      acc[c] = v;
    }
  }

  // Scatter into WMMA B-operand layout (f16).
  {
    const size_t grow = rowBase + r;            // flat row in [0, B*N)
    const int b    = (int)(grow / kN);
    const int n    = (int)(grow - (size_t)b * kN);
    const int jb   = n >> 5;
    const int jrow = n & 31;
    const int laneBase = (jrow & 16) ? 16 : 0;  // lane half-group
    const int k0       = jrow & 15;             // half index within lane
    _Float16* dp = whB + ((((size_t)b * (kN / 32) + jb) * 4 + cg) << 9)
                       + (size_t)laneBase * 16 + k0;
#pragma unroll
    for (int c = 0; c < 16; ++c) dp[c * 16] = (_Float16)acc[c];
  }

  // Attention logit projections.
  float ps = 0.f, pd = 0.f;
#pragma unroll
  for (int c = 0; c < 16; ++c) {
    ps = fmaf(acc[c], a[cb + c], ps);
    pd = fmaf(acc[c], a[kU + cb + c], pd);
  }
  sRedS[r * 4 + (t & 3)] = ps;
  sRedD[r * 4 + (t & 3)] = pd;
  __syncthreads();
  if (t < 64) {
    float s = sRedS[t * 4] + sRedS[t * 4 + 1] + sRedS[t * 4 + 2] + sRedS[t * 4 + 3];
    float d = sRedD[t * 4] + sRedD[t * 4 + 1] + sRedD[t * 4 + 2] + sRedD[t * 4 + 3];
    srcOut[rowBase + t] = s;
    dstOut[rowBase + t] = d;
  }
}

// ---------------------------------------------------------------------------
// Kernel 2: flash-style masked softmax + P@Wh via v_wmma_f32_16x16x32_f16.
// Grid: B*(N/16) blocks of 128 threads (4 wave32s). Block = 16 attention rows;
// wave cg owns U columns [cg*16, cg*16+16). 64 j-steps of width 64 (2 WMMA/step).
// ---------------------------------------------------------------------------
__global__ void __launch_bounds__(128)
gat_flash_attn_kernel(const int* __restrict__ adj,
                      const _Float16* __restrict__ whB,
                      const float* __restrict__ srcv,
                      const float* __restrict__ dstv,
                      const float* __restrict__ gamma,
                      const float* __restrict__ beta,
                      const float* __restrict__ mmean,
                      const float* __restrict__ mvar,
                      float* __restrict__ out)
{
  constexpr int EST = 72;   // sE row stride (floats); keeps float4 gathers aligned

  __shared__ alignas(16) float sE[16 * EST];   // e tile, then p tile (16x64)
  __shared__ alignas(16) float sDstAll[kN];    // all dst logits for this batch
  __shared__ float sSrc[16];
  __shared__ float sPart[16 * 8];
  __shared__ float sM[16], sS[16], sScale[16];

  const int t    = threadIdx.x;
  const int lane = t & 31;      // wave32
  const int cg   = t >> 5;      // wave id = U column group
  const int b       = blockIdx.x / (kN / 16);
  const int rowBase = (blockIdx.x % (kN / 16)) * 16;

  // Preload all dst values for this batch (coalesced, once).
  {
    const float4* dp  = (const float4*)(dstv + (size_t)b * kN);
    float4*       sdp = (float4*)sDstAll;
    for (int i = t; i < kN / 4; i += 128) sdp[i] = dp[i];
  }
  if (t < 16) {
    sM[t]   = kNegBig;
    sS[t]   = 0.f;
    sSrc[t] = srcv[(size_t)b * kN + rowBase + t];
  }

  const int ei = t >> 3;        // e-tile row this thread fills
  const int ej = (t & 7) * 8;   // e-tile col base (8 cols per thread)

  const int   col  = cg * 16 + (lane & 15);   // global U column
  const float g_   = gamma[col];
  const float be_  = beta[col];
  const float mm_  = mmean[col];
  const float inv_ = rsqrtf(mvar[col] + kBnEps);

  const int rlo  = (lane >= 16) ? 8 : 0;   // C/D row offset for this lane half
  const int klo  = (lane >= 16) ? 16 : 0;  // A-operand K offset
  const int arow = lane & 15;              // A-operand M row

  v8f acc = {0.f, 0.f, 0.f, 0.f, 0.f, 0.f, 0.f, 0.f};

  // Per-wave base of the pre-packed B-operand stream (16 halves/lane/32-row blk).
  const _Float16* whBase = whB + (((size_t)b * (kN / 32) * 4 + cg) << 9)
                               + (size_t)lane * 16;
  const size_t adjRow = ((size_t)b * kN + rowBase + ei) * kN;

  for (int j0 = 0; j0 < kN; j0 += 64) {
    // B operands: one aligned 32B global load each (L2-resident), no LDS.
    const int jb0 = j0 >> 5;
    const v16h bW0 = *(const v16h*)(whBase + (size_t)jb0 * 2048);
    const v16h bW1 = *(const v16h*)(whBase + (size_t)(jb0 + 1) * 2048);

    __syncthreads();   // previous iteration's A-gathers done; sE reusable

    // --- logits: e = mask(leakyrelu(src_i + dst_j)), 8 per thread -------
    const int4 av0 = *(const int4*)(adj + adjRow + j0 + ej);
    const int4 av1 = *(const int4*)(adj + adjRow + j0 + ej + 4);
    if (j0 + 64 < kN)
      __builtin_prefetch(adj + adjRow + j0 + 64 + ej, 0, 1);  // global_prefetch
    {
      const float si = sSrc[ei];
      float e[8];
      const int m[8] = {av0.x, av0.y, av0.z, av0.w, av1.x, av1.y, av1.z, av1.w};
#pragma unroll
      for (int q = 0; q < 8; ++q) {
        float v = si + sDstAll[j0 + ej + q];
        v = (v >= 0.f) ? v : 0.2f * v;
        e[q] = (m[q] > 0) ? v : kNegBig;
        sE[ei * EST + ej + q] = e[q];
      }
      float mx = e[0];
#pragma unroll
      for (int q = 1; q < 8; ++q) mx = fmaxf(mx, e[q]);
      sPart[ei * 8 + (t & 7)] = mx;
    }
    __syncthreads();

    // --- online max update ---------------------------------------------
    if (t < 16) {
      float bm = sPart[t * 8];
#pragma unroll
      for (int q = 1; q < 8; ++q) bm = fmaxf(bm, sPart[t * 8 + q]);
      const float mo = sM[t];
      const float mn = fmaxf(mo, bm);
      sScale[t] = __expf(mo - mn);
      sM[t] = mn;
    }
    __syncthreads();

    // --- p = exp(e - m), block row sums --------------------------------
    {
      const float m = sM[ei];
      float sum = 0.f;
#pragma unroll
      for (int q = 0; q < 8; ++q) {
        const float p = __expf(sE[ei * EST + ej + q] - m);
        sE[ei * EST + ej + q] = p;
        sum += p;
      }
      sPart[ei * 8 + (t & 7)] = sum;
    }
    __syncthreads();

    if (t < 16) {
      float bs = 0.f;
#pragma unroll
      for (int q = 0; q < 8; ++q) bs += sPart[t * 8 + q];
      sS[t] = sS[t] * sScale[t] + bs;
    }

    // --- rescale accumulator (flash correction) ------------------------
#pragma unroll
    for (int rr = 0; rr < 8; ++rr) acc[rr] *= sScale[rlo + rr];

    // --- gather A operands (P, two 16x32 tiles) per A-matrix layout -----
    const float* rowp = &sE[arow * EST];
#pragma unroll
    for (int half = 0; half < 2; ++half) {
      const float* hp = rowp + half * 32;
      const float4 f0 = *(const float4*)(hp + klo);
      const float4 f1 = *(const float4*)(hp + klo + 4);
      const float4 f2 = *(const float4*)(hp + klo + 8);
      const float4 f3 = *(const float4*)(hp + klo + 12);
      v16h aP;
      aP[0]  = (_Float16)f0.x; aP[1]  = (_Float16)f0.y;
      aP[2]  = (_Float16)f0.z; aP[3]  = (_Float16)f0.w;
      aP[4]  = (_Float16)f1.x; aP[5]  = (_Float16)f1.y;
      aP[6]  = (_Float16)f1.z; aP[7]  = (_Float16)f1.w;
      aP[8]  = (_Float16)f2.x; aP[9]  = (_Float16)f2.y;
      aP[10] = (_Float16)f2.z; aP[11] = (_Float16)f2.w;
      aP[12] = (_Float16)f3.x; aP[13] = (_Float16)f3.y;
      aP[14] = (_Float16)f3.z; aP[15] = (_Float16)f3.w;

      acc = __builtin_amdgcn_wmma_f32_16x16x32_f16(
          /*neg_a=*/false, aP, /*neg_b=*/false, (half == 0) ? bW0 : bW1,
          /*c_mod=*/(short)0, acc, /*reuse_a=*/false, /*reuse_b=*/false);
    }
  }

  __syncthreads();  // sS final

  // Epilogue: softmax normalize, BN (inference), ReLU.
#pragma unroll
  for (int rr = 0; rr < 8; ++rr) {
    const int row = rlo + rr;
    float val = acc[rr] / sS[row];
    val = (val - mm_) * inv_ * g_ + be_;
    val = fmaxf(val, 0.f);
    out[((size_t)b * kN + rowBase + row) * kU + col] = val;
  }
}

// ---------------------------------------------------------------------------
extern "C" void kernel_launch(void* const* d_in, const int* in_sizes, int n_in,
                              void* d_out, int out_size, void* d_ws, size_t ws_size,
                              hipStream_t stream) {
  (void)in_sizes; (void)n_in; (void)out_size; (void)ws_size;

  const float* h     = (const float*)d_in[0];   // (B,N,F) f32
  const int*   adj   = (const int*)d_in[1];     // (B,N,N) i32
  const float* W     = (const float*)d_in[2];   // (F,U)   f32
  const float* a     = (const float*)d_in[3];   // (2U,1)  f32
  const float* gamma = (const float*)d_in[4];
  const float* beta  = (const float*)d_in[5];
  const float* mmean = (const float*)d_in[6];
  const float* mvar  = (const float*)d_in[7];
  float* out = (float*)d_out;

  char* ws = (char*)d_ws;
  _Float16* whB  = (_Float16*)ws;                                     // 2 MB
  float*    srcv = (float*)(ws + (size_t)kB * kN * kU * sizeof(_Float16));
  float*    dstv = srcv + (size_t)kB * kN;

  gat_wh_src_dst_kernel<<<(kB * kN) / 64, 256, 0, stream>>>(
      h, W, a, whB, srcv, dstv);

  gat_flash_attn_kernel<<<kB * (kN / 16), 128, 0, stream>>>(
      adj, whB, srcv, dstv, gamma, beta, mmean, mvar, out);
}
